// PathClassifier_87256555585545
// MI455X (gfx1250) — matrix operand order repaired and checked by reading, not compiled, not run on hardware
//
#include <hip/hip_runtime.h>
#include <hip/hip_bf16.h>

typedef _Float16 half_t;
typedef __attribute__((ext_vector_type(16))) _Float16 v16h;
typedef __attribute__((ext_vector_type(8)))  float    v8f;

#define BATCH 512
#define DSIZE 512
#define HIDN  1024
#define OUTN  1000
#define OUTP  1008   // padded to multiple of 16
#define MAXM  160    // counts are in [50, 150]

// ---------------- exclusive prefix sum of counts (1 block of 512) ----------------
__global__ void scan_kernel(const int* __restrict__ counts, int* __restrict__ offsets) {
    __shared__ int sh[BATCH];
    int tid = threadIdx.x;
    sh[tid] = counts[tid];
    __syncthreads();
    for (int d = 1; d < BATCH; d <<= 1) {
        int v = (tid >= d) ? sh[tid - d] : 0;
        __syncthreads();
        sh[tid] += v;
        __syncthreads();
    }
    offsets[tid] = sh[tid] - counts[tid];
}

// ---------------- f32 -> f16 converts ----------------
__global__ void cvt_kernel(const float* __restrict__ src, half_t* __restrict__ dst, int n) {
    int i = blockIdx.x * blockDim.x + threadIdx.x;
    if (i < n) dst[i] = (half_t)src[i];
}

__global__ void cvt_pad_kernel(const float* __restrict__ src, half_t* __restrict__ dst,
                               int rows, int cols, int padRows) {
    int i = blockIdx.x * blockDim.x + threadIdx.x;
    if (i >= padRows * cols) return;
    int r = i / cols;
    dst[i] = (r < rows) ? (half_t)src[i] : (half_t)0.0f;
}

// ---------------- per-group mean -> concat16[b][0:512] ----------------
__global__ void mean_kernel(const float* __restrict__ enc, const int* __restrict__ counts,
                            const int* __restrict__ offsets, half_t* __restrict__ concat16) {
    int b   = blockIdx.x;
    int cnt = counts[b];
    long off = offsets[b];
    float inv = 1.0f / (float)cnt;
    for (int d = threadIdx.x; d < DSIZE; d += blockDim.x) {
        float acc = 0.0f;
        const float* p = enc + off * DSIZE + d;
        for (int i = 0; i < cnt; ++i) acc += p[(long)i * DSIZE];
        concat16[(long)b * (2 * DSIZE) + d] = (half_t)(acc * inv);
    }
}

// ---------------- generic WMMA GEMM: out = act(A @ Bw^T + bias) ----------------
// A: [M, lda] f16 row-major (uses K leading cols); Bw: [Npad, ldb] f16 row-major (W layout [n][k])
// One wave per 16x16 output tile. K must be a multiple of 32, M/Npad multiples of 16.
__global__ void wmma_gemm_kernel(const half_t* __restrict__ A, int lda,
                                 const half_t* __restrict__ Bw, int ldb,
                                 const float* __restrict__ bias,
                                 float* __restrict__ outF, half_t* __restrict__ outH, int ldo,
                                 int M, int Npad, int Nstore, int K, int doTanh) {
    int wave = (blockIdx.x * blockDim.x + threadIdx.x) >> 5;
    int lane = threadIdx.x & 31;
    int tilesN = Npad >> 4;
    int totalTiles = (M >> 4) * tilesN;
    if (wave >= totalTiles) return;          // wave-uniform: EXEC stays all-1s for WMMA

    int tm = wave / tilesN;
    int tn = wave % tilesN;
    int h  = lane >> 4;                      // K-half select
    int ml = lane & 15;                      // A: row-in-tile; B/D: col-in-tile
    long rowA = (long)(tm * 16 + ml) * lda;
    long colB = (long)(tn * 16 + ml) * ldb;

    v8f acc = {};
#pragma unroll 4
    for (int kb = 0; kb < K; kb += 32) {
        v16h a, bfr;
#pragma unroll
        for (int j = 0; j < 8; ++j) {
            // A 16x32 layout: lanes 0-15 VGPR j<4 -> K=2j..; j>=4 -> K=16+..; lanes 16-31 offset +8
            int ka = kb + ((j < 4) ? (2 * j) : (16 + 2 * (j - 4))) + h * 8;
            a[2 * j]     = A[rowA + ka];
            a[2 * j + 1] = A[rowA + ka + 1];
            // B 32x16 layout: same K mapping as A (j<4 -> K=2j..; j>=4 -> K=16+..; lanes 16-31 offset +8)
            int kbb = kb + ((j < 4) ? (2 * j) : (16 + 2 * (j - 4))) + h * 8;
            bfr[2 * j]     = Bw[colB + kbb];
            bfr[2 * j + 1] = Bw[colB + kbb + 1];
        }
        acc = __builtin_amdgcn_wmma_f32_16x16x32_f16(false, a, false, bfr,
                                                     (short)0, acc, false, false);
    }

    int n = tn * 16 + ml;
#pragma unroll
    for (int r = 0; r < 8; ++r) {
        int m = tm * 16 + r + 8 * h;         // D layout: VGPR r -> M = r + 8*half
        if (n < Nstore) {
            float v = acc[r];
            if (bias)   v += bias[n];
            if (doTanh) v = tanhf(v);
            if (outF) outF[(long)m * ldo + n] = v;
            if (outH) outH[(long)m * ldo + n] = (half_t)v;
        }
    }
}

// ---------------- fused Luong attention per group ----------------
__global__ void attn_kernel(const float* __restrict__ enc, const int* __restrict__ counts,
                            const int* __restrict__ offsets, const float* __restrict__ q,
                            half_t* __restrict__ concat16) {
    __shared__ float qsh[DSIZE];
    __shared__ float sc[MAXM];
    __shared__ float red[256];
    int b   = blockIdx.x;
    int tid = threadIdx.x;
    int cnt = counts[b];
    long off = offsets[b];

    for (int d = tid; d < DSIZE; d += blockDim.x) qsh[d] = q[(long)b * DSIZE + d];
    __syncthreads();

    // scores: one row per wave, 16 dims per lane, shfl reduce
    int lane = tid & 31, wv = tid >> 5, nw = blockDim.x >> 5;
    for (int i = wv; i < cnt; i += nw) {
        const float* row = enc + (off + i) * DSIZE;
        float p = 0.0f;
        int d0 = lane * 16;
#pragma unroll
        for (int u = 0; u < 16; ++u) p += row[d0 + u] * qsh[d0 + u];
        for (int s = 16; s > 0; s >>= 1) p += __shfl_down(p, s);
        if (lane == 0) sc[i] = p;
    }
    __syncthreads();

    // softmax (stable)
    float lm = -1e30f;
    for (int i = tid; i < cnt; i += blockDim.x) lm = fmaxf(lm, sc[i]);
    red[tid] = lm; __syncthreads();
    for (int s = 128; s > 0; s >>= 1) {
        if (tid < s) red[tid] = fmaxf(red[tid], red[tid + s]);
        __syncthreads();
    }
    float mx = red[0]; __syncthreads();

    float ls = 0.0f;
    for (int i = tid; i < cnt; i += blockDim.x) {
        float e = expf(sc[i] - mx);
        sc[i] = e;
        ls += e;
    }
    red[tid] = ls; __syncthreads();
    for (int s = 128; s > 0; s >>= 1) {
        if (tid < s) red[tid] += red[tid + s];
        __syncthreads();
    }
    float inv = 1.0f / red[0];
    __syncthreads();

    // context = attn^T @ rows -> concat16[b][512:1024]
    for (int d = tid; d < DSIZE; d += blockDim.x) {
        float acc = 0.0f;
        const float* p = enc + off * DSIZE + d;
        for (int i = 0; i < cnt; ++i) acc += sc[i] * p[(long)i * DSIZE];
        concat16[(long)b * (2 * DSIZE) + DSIZE + d] = (half_t)(acc * inv);
    }
}

extern "C" void kernel_launch(void* const* d_in, const int* in_sizes, int n_in,
                              void* d_out, int out_size, void* d_ws, size_t ws_size,
                              hipStream_t stream) {
    (void)in_sizes; (void)n_in; (void)out_size; (void)ws_size;
    const float* enc      = (const float*)d_in[0];
    const int*   counts   = (const int*)d_in[1];
    const float* W_attn   = (const float*)d_in[2];
    const float* W_concat = (const float*)d_in[3];
    const float* b_concat = (const float*)d_in[4];
    const float* W_cls    = (const float*)d_in[5];
    const float* b_cls    = (const float*)d_in[6];
    float* out = (float*)d_out;

    char* ws = (char*)d_ws;
    size_t p = 0;
    auto alloc = [&](size_t bytes) -> char* {
        p = (p + 255) & ~(size_t)255;
        char* r = ws + p;
        p += bytes;
        return r;
    };
    int*    offsets   = (int*)alloc(BATCH * sizeof(int));
    half_t* concat16  = (half_t*)alloc((size_t)BATCH * 2 * DSIZE * sizeof(half_t));
    float*  qbuf      = (float*)alloc((size_t)BATCH * DSIZE * sizeof(float));
    half_t* wattn16   = (half_t*)alloc((size_t)DSIZE * DSIZE * sizeof(half_t));
    half_t* wconcat16 = (half_t*)alloc((size_t)HIDN * 2 * DSIZE * sizeof(half_t));
    half_t* wcls16    = (half_t*)alloc((size_t)OUTP * HIDN * sizeof(half_t));
    half_t* h16       = (half_t*)alloc((size_t)BATCH * HIDN * sizeof(half_t));

    // 1) offsets
    scan_kernel<<<1, BATCH, 0, stream>>>(counts, offsets);
    // 2) weight converts
    cvt_kernel<<<(DSIZE * DSIZE + 255) / 256, 256, 0, stream>>>(W_attn, wattn16, DSIZE * DSIZE);
    cvt_kernel<<<(HIDN * 2 * DSIZE + 255) / 256, 256, 0, stream>>>(W_concat, wconcat16, HIDN * 2 * DSIZE);
    cvt_pad_kernel<<<(OUTP * HIDN + 255) / 256, 256, 0, stream>>>(W_cls, wcls16, OUTN, HIDN, OUTP);
    // 3) mean pooling
    mean_kernel<<<BATCH, 256, 0, stream>>>(enc, counts, offsets, concat16);
    // 4) q = initial @ W_attn^T   (A = first 512 cols of concat16, lda = 1024)
    {
        int tiles = (BATCH / 16) * (DSIZE / 16);
        wmma_gemm_kernel<<<(tiles + 7) / 8, 256, 0, stream>>>(
            concat16, 2 * DSIZE, wattn16, DSIZE, nullptr,
            qbuf, nullptr, DSIZE, BATCH, DSIZE, DSIZE, DSIZE, 0);
    }
    // 5) attention -> concat16[b][512:1024]
    attn_kernel<<<BATCH, 256, 0, stream>>>(enc, counts, offsets, qbuf, concat16);
    // 6) H = tanh(concat @ W_concat^T + b_concat) -> f16
    {
        int tiles = (BATCH / 16) * (HIDN / 16);
        wmma_gemm_kernel<<<(tiles + 7) / 8, 256, 0, stream>>>(
            concat16, 2 * DSIZE, wconcat16, 2 * DSIZE, b_concat,
            nullptr, h16, HIDN, BATCH, HIDN, HIDN, 2 * DSIZE, 1);
    }
    // 7) out = H @ W_cls^T + b_cls
    {
        int tiles = (BATCH / 16) * (OUTP / 16);
        wmma_gemm_kernel<<<(tiles + 7) / 8, 256, 0, stream>>>(
            h16, HIDN, wcls16, HIDN, b_cls,
            out, nullptr, OUTN, BATCH, OUTP, OUTN, HIDN, 0);
    }
}
